// FFM_67482526155458
// MI455X (gfx1250) — matrix operand order repaired and checked by reading, not compiled, not run on hardware
//
#include <hip/hip_runtime.h>
#include <hip/hip_bf16.h>
#include <math.h>

// Problem constants (match reference)
#define T_STEPS 4096
#define IN_DIM  512
#define OUT_DIM 512
#define M_DIM   256
#define C_DIM   16
#define MC      (M_DIM * C_DIM)   // 4096 state elements
#define ZK      (2 * MC)          // 8192 = mix GEMM K
#define CHUNK   64
#define NCHUNK  (T_STEPS / CHUNK) // 64 chunks
#define KSLAB   1024              // bf16 K columns per TDM slab
#define NSLAB   (ZK / KSLAB)      // 8 slabs
#define SLAB_BYTES (16 * KSLAB * 2)  // 32 KB per buffer

typedef __attribute__((ext_vector_type(2)))  float        v2f;
typedef __attribute__((ext_vector_type(8)))  float        v8f;
typedef __attribute__((ext_vector_type(16))) __bf16       v16bf;
typedef __attribute__((ext_vector_type(4)))  unsigned int v4u;
typedef __attribute__((ext_vector_type(8)))  int          v8i;
typedef __attribute__((ext_vector_type(4)))  int          v4i;

__device__ __forceinline__ v8f wmma_f32_k4(v2f a, v2f b, v8f c) {
  return __builtin_amdgcn_wmma_f32_16x16x4_f32(false, a, false, b, (short)0, c,
                                               false, false);
}
__device__ __forceinline__ v8f wmma_bf16_k32(v16bf a, v16bf b, v8f c) {
  return __builtin_amdgcn_wmma_f32_16x16x32_bf16(false, a, false, b, (short)0, c,
                                                 false, false);
}

// ---------------------------------------------------------------------------
// TDM: 2-D tile DMA, global (bf16, row stride in elems) -> LDS, per
// CDNA5 ISA. Wave-level op; EXEC ignored; TENSORcnt.
__device__ __forceinline__ void tdm_load_2d_bf16(unsigned int lds_off,
                                                 const void* gaddr,
                                                 unsigned int tile_w,
                                                 unsigned int tile_h,
                                                 unsigned int row_stride) {
  const unsigned long long ga = (unsigned long long)gaddr;
  v4u g0;
  g0[0] = 1u;                                   // count=1, user descriptor
  g0[1] = lds_off;                              // LDS byte address
  g0[2] = (unsigned int)(ga & 0xFFFFFFFFu);     // global_addr[31:0]
  g0[3] = (unsigned int)((ga >> 32) & 0x01FFFFFFu) | 0x80000000u; // type=2
  const unsigned int td0 = 1u << 24;            // generous tensor extents
  const unsigned int td1 = 1u << 20;            // (tile never goes OOB)
  v8i g1;
  g1[0] = (int)(1u << 16);                      // data_size=1 (2 bytes)
  g1[1] = (int)((td0 & 0xFFFFu) << 16);         // tensor_dim0[15:0] @ bits63:48
  g1[2] = (int)((td0 >> 16) | ((td1 & 0xFFFFu) << 16));
  g1[3] = (int)((td1 >> 16) | (tile_w << 16));  // tile_dim0 @ bits127:112
  g1[4] = (int)(tile_h & 0xFFFFu);              // tile_dim1 (tile_dim2=0)
  g1[5] = (int)row_stride;                      // tensor_dim0_stride[31:0]
  g1[6] = 0;
  g1[7] = 0;
  v4i g2 = {0, 0, 0, 0};
  v4i g3 = {0, 0, 0, 0};
#if __has_include(<hip/amd_detail/amd_gfx1250_TDM.h>)
  v8i g4 = {0, 0, 0, 0, 0, 0, 0, 0};            // therock toolchain: 6-arg form
  __builtin_amdgcn_tensor_load_to_lds(g0, g1, g2, g3, g4, 0);
#else
  __builtin_amdgcn_tensor_load_to_lds(g0, g1, g2, g3, 0);  // ROCm 7.2: 5-arg
#endif
}

// ---------------------------------------------------------------------------
// mix_w f32 -> bf16 (row-major; bf16 B-operand per-lane K run is contiguous)
__global__ void cvt_mixw_bf16(const float* __restrict__ w,
                              __bf16* __restrict__ wb) {
  const size_t i = ((size_t)blockIdx.x * blockDim.x + threadIdx.x) * 4;
#pragma unroll
  for (int j = 0; j < 4; ++j) wb[i + j] = (__bf16)w[i + j];
}

// ---------------------------------------------------------------------------
// pre = L2-row-normalize(x @ pre_w^T + pre_b)   [T, M]
__global__ __launch_bounds__(512) void pre_gemm_norm(
    const float* __restrict__ x, const float* __restrict__ pre_w,
    const float* __restrict__ pre_b, float* __restrict__ pre_out) {
  __shared__ float lds[16 * M_DIM];
  __shared__ float norms[16];
  const int tid  = threadIdx.x;
  const int wave = tid >> 5, lane = tid & 31;
  const int half = lane >> 4, ln = lane & 15;
  const int t0 = blockIdx.x * 16;
  const int m0 = wave * 16;

  v8f acc = {0.f, 0.f, 0.f, 0.f, 0.f, 0.f, 0.f, 0.f};
  const float* ax = x + (size_t)(t0 + ln) * IN_DIM + half * 2;
  const float* bw = pre_w + (size_t)(m0 + ln) * IN_DIM + half * 2;
  for (int k = 0; k < IN_DIM; k += 4) {
    v2f a = *(const v2f*)(ax + k);
    v2f b = *(const v2f*)(bw + k);
    acc = wmma_f32_k4(a, b, acc);
  }
  const float bias = pre_b[m0 + ln];
#pragma unroll
  for (int r = 0; r < 8; ++r) {
    lds[(r + 8 * half) * M_DIM + m0 + ln] = acc[r] + bias;
  }
  __syncthreads();
  if (tid < 16) norms[tid] = 0.f;
  __syncthreads();
  if (tid < 256) {
    const int row = tid >> 4, seg = tid & 15;
    float p = 0.f;
#pragma unroll
    for (int j = 0; j < 16; ++j) {
      const float v = lds[row * M_DIM + seg * 16 + j];
      p += v * v;
    }
    atomicAdd(&norms[row], p);  // ds_add_f32
  }
  __syncthreads();
  for (int e = tid; e < 16 * M_DIM; e += 512) {
    const int row = e >> 8;
    const float inv = 1.f / (1e-6f + sqrtf(norms[row]));
    pre_out[(size_t)(t0 + row) * M_DIM + (e & 255)] = lds[e] * inv;
  }
}

// ---------------------------------------------------------------------------
// Chunked affine scan: s_t = A_t * s_{t-1} + p_t,  A_t = (start? 0 : d_mc)
__global__ void ffa_scan_chunks(const float* __restrict__ pre,
                                const unsigned char* __restrict__ start,
                                const float* __restrict__ ffa_a,
                                const float* __restrict__ ffa_b,
                                float* __restrict__ chunkSum) {
  const int idx = blockIdx.y * blockDim.x + threadIdx.x;  // 0..4095
  const int m = idx >> 4, c = idx & 15;
  const int g = blockIdx.x;
  const float er = __expf(-fabsf(ffa_a[m]));
  const float ph = ffa_b[c];
  const float dr = er * __cosf(ph), di = er * __sinf(ph);
  float sr = 0.f, si = 0.f, ar = 1.f, ai = 0.f;
  const int tend = (g + 1) * CHUNK;
  for (int t = g * CHUNK; t < tend; ++t) {
    const float p  = pre[(size_t)t * M_DIM + m];
    const float rs = start[t] ? 0.f : 1.f;
    float nr = rs * (sr * dr - si * di) + p;
    float ni = rs * (sr * di + si * dr);
    sr = nr; si = ni;
    nr = rs * (ar * dr - ai * di);
    ni = rs * (ar * di + ai * dr);
    ar = nr; ai = ni;
  }
  float* o = chunkSum + ((size_t)g * MC + idx) * 4;
  o[0] = sr; o[1] = si; o[2] = ar; o[3] = ai;
}

__global__ void ffa_scan_heads(const float* __restrict__ state0,
                               const float* __restrict__ chunkSum,
                               float* __restrict__ inits) {
  const int idx = blockIdx.x * blockDim.x + threadIdx.x;
  float sr = state0[idx], si = 0.f;
  for (int g = 0; g < NCHUNK; ++g) {
    float* o = inits + ((size_t)g * MC + idx) * 2;
    o[0] = sr; o[1] = si;
    const float* s = chunkSum + ((size_t)g * MC + idx) * 4;
    const float ls = s[0], li = s[1], ar = s[2], ai = s[3];
    const float nr = ar * sr - ai * si + ls;
    const float ni = ar * si + ai * sr + li;
    sr = nr; si = ni;
  }
}

// Pass 3: re-run chunks from true inits; emit f32 complex state (NT stores —
// written once, never re-read; keeps zin L2-resident) + bf16 copy permuted
// into 16 re then 16 im per m (each out_gemm lane reads 16 contiguous k).
__global__ void ffa_scan_emit(const float* __restrict__ pre,
                              const unsigned char* __restrict__ start,
                              const float* __restrict__ ffa_a,
                              const float* __restrict__ ffa_b,
                              const float* __restrict__ inits,
                              float* __restrict__ state_out,
                              __bf16* __restrict__ zin) {
  const int idx = blockIdx.y * blockDim.x + threadIdx.x;
  const int m = idx >> 4, c = idx & 15;
  const int g = blockIdx.x;
  const float er = __expf(-fabsf(ffa_a[m]));
  const float ph = ffa_b[c];
  const float dr = er * __cosf(ph), di = er * __sinf(ph);
  const float* iv = inits + ((size_t)g * MC + idx) * 2;
  float sr = iv[0], si = iv[1];
  const int q0 = c;  // real slot; imag at q0+16
  const int tend = (g + 1) * CHUNK;
  for (int t = g * CHUNK; t < tend; ++t) {
    const float p  = pre[(size_t)t * M_DIM + m];
    const float rs = start[t] ? 0.f : 1.f;
    const float nr = rs * (sr * dr - si * di) + p;
    const float ni = rs * (sr * di + si * dr);
    sr = nr; si = ni;
    float* so = state_out + 2 * ((size_t)t * MC + idx);  // complex64 layout
    __builtin_nontemporal_store(sr, so);
    __builtin_nontemporal_store(si, so + 1);
    __bf16* z = zin + (size_t)t * ZK + m * 32;
    z[q0]     = (__bf16)sr;
    z[q0 + 16] = (__bf16)si;
  }
}

// ---------------------------------------------------------------------------
// out = zin(bf16) @ mix_w^T + x @ skip_w^T + biases.
// One 1024-thread block (32 waves) per 16-row block; wave w owns col tile
// m0 = 16w. Shared A slab (16 x KSLAB bf16) staged by TDM into a double-
// buffered LDS region; wave 0 issues tensor_load_to_lds, synchronizes via
// s_wait_tensorcnt + workgroup barriers. B streams from global (L2-resident).
__global__ __launch_bounds__(1024) void out_gemm(
    const __bf16* __restrict__ zin, const __bf16* __restrict__ mixw_bf,
    const float* __restrict__ x, const float* __restrict__ skip_w,
    const float* __restrict__ mix_b, const float* __restrict__ skip_b,
    float* __restrict__ out) {
  __shared__ __align__(32) __bf16 asl[2][16 * KSLAB];  // 64 KB double buffer
  const int tid  = threadIdx.x;
  const int wave = tid >> 5, lane = tid & 31;
  const int half = lane >> 4, ln = lane & 15;
  const int t0 = blockIdx.x * 16;
  const int m0 = wave * 16;

  const unsigned int lds_base =
      (unsigned int)(unsigned long long)(void*)&asl[0][0];
  const __bf16* agbase = zin + (size_t)t0 * ZK;

  if (wave == 0) {  // prologue: slab 0 in flight
    tdm_load_2d_bf16(lds_base, agbase, KSLAB, 16, ZK);
  }

  v8f acc = {0.f, 0.f, 0.f, 0.f, 0.f, 0.f, 0.f, 0.f};
  const __bf16* wb = mixw_bf + (size_t)(m0 + ln) * ZK + half * 16;

  for (int s = 0; s < NSLAB; ++s) {
    if (wave == 0) {
      if (s + 1 < NSLAB) {
        tdm_load_2d_bf16(lds_base + ((s + 1) & 1) * SLAB_BYTES,
                         agbase + (size_t)(s + 1) * KSLAB, KSLAB, 16, ZK);
        __builtin_amdgcn_s_wait_tensorcnt(1);   // slab s landed
      } else {
        __builtin_amdgcn_s_wait_tensorcnt(0);   // last slab landed
      }
    }
    __syncthreads();  // slab s visible to all waves
    const __bf16* as = &asl[s & 1][ln * KSLAB + half * 16];
    const __bf16* wbs = wb + (size_t)s * KSLAB;
#pragma unroll 8
    for (int kb = 0; kb < KSLAB; kb += 32) {
      v16bf a = *(const v16bf*)(as + kb);       // ds_load
      v16bf b = *(const v16bf*)(wbs + kb);      // global_load
      acc = wmma_bf16_k32(a, b, acc);
    }
    __syncthreads();  // everyone done with buffer (s&1) before slab s+2 DMA
  }

  // Fused skip GEMM in full f32 WMMA (K=512)
  const float* xa = x + (size_t)(t0 + ln) * IN_DIM + half * 2;
  const float* sw = skip_w + (size_t)(m0 + ln) * IN_DIM + half * 2;
  for (int k = 0; k < IN_DIM; k += 4) {
    v2f a = *(const v2f*)(xa + k);
    v2f b = *(const v2f*)(sw + k);
    acc = wmma_f32_k4(a, b, acc);
  }
  const int col = m0 + ln;
  const float bias = mix_b[col] + skip_b[col];
#pragma unroll
  for (int r = 0; r < 8; ++r) {
    __builtin_nontemporal_store(
        acc[r] + bias, &out[(size_t)(t0 + r + 8 * half) * OUT_DIM + col]);
  }
}

// ---------------------------------------------------------------------------
extern "C" void kernel_launch(void* const* d_in, const int* in_sizes, int n_in,
                              void* d_out, int out_size, void* d_ws,
                              size_t ws_size, hipStream_t stream) {
  (void)in_sizes; (void)n_in; (void)out_size; (void)ws_size;
  const float* x      = (const float*)d_in[0];
  const float* state0 = (const float*)d_in[1];
  const unsigned char* start = (const unsigned char*)d_in[2];  // bool8
  const float* pre_w  = (const float*)d_in[4];
  const float* pre_b  = (const float*)d_in[5];
  const float* mix_w  = (const float*)d_in[6];
  const float* mix_b  = (const float*)d_in[7];
  const float* skip_w = (const float*)d_in[8];
  const float* skip_b = (const float*)d_in[9];
  const float* ffa_a  = (const float*)d_in[10];
  const float* ffa_b  = (const float*)d_in[11];

  float* out       = (float*)d_out;                        // [T, OUT] f32
  float* state_out = out + (size_t)T_STEPS * OUT_DIM;      // [T,M,C] complex64

  char* ws = (char*)d_ws;
  float* pre      = (float*)ws;  ws += (size_t)T_STEPS * M_DIM * 4;   //  4 MB
  float* chunkSum = (float*)ws;  ws += (size_t)NCHUNK * MC * 4 * 4;   //  4 MB
  float* inits    = (float*)ws;  ws += (size_t)NCHUNK * MC * 2 * 4;   //  2 MB
  __bf16* mixw_bf = (__bf16*)ws; ws += (size_t)OUT_DIM * ZK * 2;      //  8 MB
  __bf16* zin     = (__bf16*)ws;                                      // 64 MB

  cvt_mixw_bf16<<<(OUT_DIM * ZK) / 1024, 256, 0, stream>>>(mix_w, mixw_bf);
  pre_gemm_norm<<<T_STEPS / 16, 512, 0, stream>>>(x, pre_w, pre_b, pre);
  ffa_scan_chunks<<<dim3(NCHUNK, MC / 256), 256, 0, stream>>>(pre, start, ffa_a,
                                                              ffa_b, chunkSum);
  ffa_scan_heads<<<MC / 256, 256, 0, stream>>>(state0, chunkSum, inits);
  ffa_scan_emit<<<dim3(NCHUNK, MC / 256), 256, 0, stream>>>(
      pre, start, ffa_a, ffa_b, inits, state_out, zin);
  out_gemm<<<T_STEPS / 16, 1024, 0, stream>>>(zin, mixw_bf, x, skip_w, mix_b,
                                              skip_b, out);
}
